// DynamicSparseAttention_82892868813321
// MI455X (gfx1250) — matrix operand order repaired and checked by reading, not compiled, not run on hardware
//
#include <hip/hip_runtime.h>
#include <hip/hip_bf16.h>

// ---------------------------------------------------------------------------
// Dynamic sparse attention, CDNA5 (gfx1250) WMMA bf16 pipeline, v2.
// All matrix operands live in memory in WMMA fragment-swizzled order
// ([kstep][frag][lane][16 bf16]) so every fragment load is 2 x b128.
// No LDS in the dense GEMMs; scores never materialized (recomputed in the
// histogram / row-stats / accumulate passes via v_wmma_f32_16x16x32_bf16).
// ---------------------------------------------------------------------------

typedef __attribute__((ext_vector_type(16))) __bf16 v16bf;
typedef __attribute__((ext_vector_type(8)))  float  v8f;

constexpr int   kS  = 2048;
constexpr int   kD  = 1024;
constexpr int   kH  = 16;
constexpr int   kHD = 64;
constexpr float kScale = 0.125f;          // 1/sqrt(64)
constexpr int   NBINS  = 1024;
constexpr float SMIN   = -8.0f;
constexpr float SMAX   =  8.0f;
constexpr float BW     = (SMAX - SMIN) / (float)NBINS;

// ---------------- WMMA + fragment swizzle helpers --------------------------

__device__ inline v8f wmma_bf16(v16bf a, v16bf b, v8f c) {
  return __builtin_amdgcn_wmma_f32_16x16x32_bf16(
      /*neg_a=*/false, a, /*neg_b=*/false, b,
      /*c_mod=*/(short)0, c, /*reuse_a=*/false, /*reuse_b=*/false);
}

// A fragment (16x32 MxK bf16) lane layout (contiguous k order used by this code, not the ISA table order):
//   lane = m + 16*(kk>>4), elem = kk & 15
__device__ inline int a_swz(int m, int kk) {
  return (m + ((kk >> 4) << 4)) * 16 + (kk & 15);
}
// B fragment (32x16 KxN bf16) lane layout:
//   lane = n + 16*(kk>>4), elem = kk & 15
__device__ inline int b_swz(int kk, int nn) {
  return (nn + ((kk >> 4) << 4)) * 16 + (kk & 15);
}

// Load one fragment: lane reads its 16 contiguous bf16 (32B) = 2 x b128.
__device__ inline v16bf load_frag(const __bf16* __restrict__ base) {
  const int lane = threadIdx.x & 31;
  union { float4 f[2]; v16bf v; } u;
  const float4* p = reinterpret_cast<const float4*>(base + lane * 16);
  u.f[0] = p[0];
  u.f[1] = p[1];
  return u.v;
}

// ---------------- fp32 -> bf16 swizzling conversions ------------------------

// x [2048][1024] -> A-frag order: dst[((ks*128 + mf)<<9) + a_swz(m,kk)]
__global__ void cvt_x_swzA(const float* __restrict__ src,
                           __bf16* __restrict__ dst) {
  int i = blockIdx.x * 256 + threadIdx.x;
  if (i >= kS * kD) return;
  const int row = i >> 10, col = i & 1023;
  const int ks = col >> 5, kk = col & 31;
  const int mf = row >> 4, m = row & 15;
  dst[(((size_t)ks * 128 + mf) << 9) + a_swz(m, kk)] = (__bf16)src[i];
}

// W [1024][1024] (k x n) -> B-frag order: dst[((ks*64 + nf)<<9) + b_swz(kk,nn)]
__global__ void cvt_w_swzB(const float* __restrict__ src,
                           __bf16* __restrict__ dst) {
  int i = blockIdx.x * 256 + threadIdx.x;
  if (i >= kD * kD) return;
  const int row = i >> 10, col = i & 1023;  // row = k, col = n
  const int ks = row >> 5, kk = row & 31;
  const int nf = col >> 4, nn = col & 15;
  dst[(((size_t)ks * 64 + nf) << 9) + b_swz(kk, nn)] = (__bf16)src[i];
}

// ---------------- dense GEMM (no LDS): C[2048x1024] = A @ W + bias ----------
// A in A-frag order, W in B-frag order. Output modes:
//   0: Q  -> [h][ks=d/32][qf][frag] A-frag order for score passes
//   1: K  -> [h][ks=d/32][kf][frag] B-frag (K^T) order for score passes
//   2: V  -> [h][ks=key/32][nf=d/16][frag] B-frag order for attn@V
//   3: fp32 row-major [S][D] (final output)

__global__ __launch_bounds__(256) void gemm_swz(
    const __bf16* __restrict__ A,    // A-frag order, 128 frags per k-step
    const __bf16* __restrict__ Bw,   // B-frag order, 64 frags per k-step
    const float*  __restrict__ bias,
    __bf16* __restrict__ outB,
    float*  __restrict__ outF,
    int mode) {
  const int tid  = threadIdx.x;
  const int wid  = tid >> 5;
  const int lane = tid & 31;
  const int half = lane >> 4;
  const int ln   = lane & 15;
  const int rowBase = blockIdx.x * 128;
  const int colBase = blockIdx.y * 128;
  const int waveM = wid & 3;   // 4 row groups of 32
  const int waveN = wid >> 2;  // 2 col groups of 64
  const int mf0 = (rowBase >> 4) + waveM * 2;
  const int nf0 = (colBase >> 4) + waveN * 4;

  v8f acc[2][4];
  v8f zero = {};
#pragma unroll
  for (int i = 0; i < 2; ++i)
#pragma unroll
    for (int j = 0; j < 4; ++j) acc[i][j] = zero;

  for (int s = 0; s < 32; ++s) {  // 32 k-steps of 32
    v16bf af[2], bf[4];
#pragma unroll
    for (int i = 0; i < 2; ++i)
      af[i] = load_frag(A + (((size_t)s * 128 + mf0 + i) << 9));
#pragma unroll
    for (int j = 0; j < 4; ++j)
      bf[j] = load_frag(Bw + (((size_t)s * 64 + nf0 + j) << 9));
    if (s + 1 < 32) {  // global_prefetch_b8 of next A fragment
      __builtin_prefetch(A + (((size_t)(s + 1) * 128 + mf0) << 9) + lane * 16,
                         0, 1);
    }
#pragma unroll
    for (int i = 0; i < 2; ++i)
#pragma unroll
      for (int j = 0; j < 4; ++j) acc[i][j] = wmma_bf16(af[i], bf[j], acc[i][j]);
  }

#pragma unroll
  for (int i = 0; i < 2; ++i)
#pragma unroll
    for (int j = 0; j < 4; ++j) {
      const int col  = colBase + waveN * 64 + j * 16 + ln;
      const float bc = bias[col];
#pragma unroll
      for (int r = 0; r < 8; ++r) {
        const int row = rowBase + waveM * 32 + i * 16 + r + 8 * half;
        const float v = acc[i][j][r] + bc;
        if (mode == 3) {
          outF[(size_t)row * kD + col] = v;
        } else {
          const int h = col >> 6, d = col & 63;
          size_t idx;
          if (mode == 0) {        // Q: A-frag of (q, d)
            idx = (((size_t)(h * 2 + (d >> 5)) * 128 + (row >> 4)) << 9) +
                  a_swz(row & 15, d & 31);
          } else if (mode == 1) { // K: B-frag of (k=d, n=key)
            idx = (((size_t)(h * 2 + (d >> 5)) * 128 + (row >> 4)) << 9) +
                  b_swz(d & 31, row & 15);
          } else {                // V: B-frag of (k=key, n=d)
            idx = (((size_t)(h * 64 + (row >> 5)) * 4 + ((d >> 4) & 3)) << 9) +
                  b_swz(row & 31, d & 15);
          }
          outB[idx] = (__bf16)v;
        }
      }
    }
}

// ---------------- per-head score histogram ---------------------------------
// grid (S/64, S/64, H), 256 threads; each wave computes a 16x32 score tile.

__global__ __launch_bounds__(256) void hist_scores(
    const __bf16* __restrict__ Q, const __bf16* __restrict__ K,
    unsigned int* __restrict__ ghist) {
  __shared__ unsigned int shist[NBINS];
  const int tid = threadIdx.x;
  for (int i = tid; i < NBINS; i += 256) shist[i] = 0u;
  __syncthreads();

  const int h  = blockIdx.z;
  const int q0 = blockIdx.y * 64;
  const int c0 = blockIdx.x * 64;
  const int wid = tid >> 5;
  const int waveM = wid & 3, waveN = wid >> 2;
  const int qf = (q0 >> 4) + waveM;

  const v16bf a0 = load_frag(Q + (((size_t)(h * 2 + 0) * 128 + qf) << 9));
  const v16bf a1 = load_frag(Q + (((size_t)(h * 2 + 1) * 128 + qf) << 9));

#pragma unroll
  for (int t = 0; t < 2; ++t) {
    const int kf = ((c0 + waveN * 32) >> 4) + t;
    const v16bf b0 = load_frag(K + (((size_t)(h * 2 + 0) * 128 + kf) << 9));
    const v16bf b1 = load_frag(K + (((size_t)(h * 2 + 1) * 128 + kf) << 9));
    v8f s = {};
    s = wmma_bf16(a0, b0, s);
    s = wmma_bf16(a1, b1, s);
#pragma unroll
    for (int r = 0; r < 8; ++r) {
      const float v = s[r] * kScale;
      int bin = (int)((v - SMIN) * (1.0f / BW));
      bin = bin < 0 ? 0 : (bin >= NBINS ? NBINS - 1 : bin);
      atomicAdd(&shist[bin], 1u);
    }
  }
  __syncthreads();
  for (int i = tid; i < NBINS; i += 256)
    if (shist[i]) atomicAdd(&ghist[h * NBINS + i], shist[i]);
}

// ---------------- histogram -> per-head 0.95 quantile threshold ------------

__global__ void find_thr(const unsigned int* __restrict__ ghist,
                         float* __restrict__ thr) {
  const int h = blockIdx.x;
  if (threadIdx.x == 0) {
    const unsigned int* hh = ghist + h * NBINS;
    const unsigned long long target =
        (unsigned long long)(0.95 * (double)kS * (double)kS);
    unsigned long long cum = 0;
    int b = NBINS - 1;
    for (int i = 0; i < NBINS; ++i) {
      cum += hh[i];
      if (cum >= target) { b = i; break; }
    }
    thr[h] = SMIN + b * BW;  // lower edge of crossing bin
  }
}

// ---------------- pass 1: per-row masked max / sum-exp ---------------------
// grid (S/64, H), 128 threads (4 waves); wave owns 16 query rows.

__global__ __launch_bounds__(128) void row_stats(
    const __bf16* __restrict__ Q, const __bf16* __restrict__ K,
    const float* __restrict__ thrv, float* __restrict__ rowm,
    float* __restrict__ rowl) {
  const int h   = blockIdx.y;
  const int q0  = blockIdx.x * 64;
  const int wid = threadIdx.x >> 5;
  const int lane = threadIdx.x & 31;
  const int half = lane >> 4, ln = lane & 15;
  const float thr = thrv[h];
  const int qrow  = q0 + wid * 16;
  const int qf    = qrow >> 4;

  const v16bf a0 = load_frag(Q + (((size_t)(h * 2 + 0) * 128 + qf) << 9));
  const v16bf a1 = load_frag(Q + (((size_t)(h * 2 + 1) * 128 + qf) << 9));

  float m[8], l[8];
#pragma unroll
  for (int r = 0; r < 8; ++r) { m[r] = -__builtin_inff(); l[r] = 0.0f; }

  for (int kb = 0; kb < kS; kb += 32) {
    v8f s[2];
#pragma unroll
    for (int t = 0; t < 2; ++t) {
      const int kf = (kb >> 4) + t;
      const v16bf b0 = load_frag(K + (((size_t)(h * 2 + 0) * 128 + kf) << 9));
      const v16bf b1 = load_frag(K + (((size_t)(h * 2 + 1) * 128 + kf) << 9));
      v8f acc = {};
      acc = wmma_bf16(a0, b0, acc);
      acc = wmma_bf16(a1, b1, acc);
      s[t] = acc;
    }
#pragma unroll
    for (int r = 0; r < 8; ++r) {
      float v0 = s[0][r] * kScale;
      float v1 = s[1][r] * kScale;
      v0 = (v0 >= thr) ? v0 : -__builtin_inff();
      v1 = (v1 >= thr) ? v1 : -__builtin_inff();
      float tmax = fmaxf(v0, v1);
#pragma unroll
      for (int off = 1; off < 16; off <<= 1)
        tmax = fmaxf(tmax, __shfl_xor(tmax, off, 16));
      const float newm = fmaxf(m[r], tmax);
      if (newm != -__builtin_inff()) {
        float p = __expf(v0 - newm) + __expf(v1 - newm);
#pragma unroll
        for (int off = 1; off < 16; off <<= 1) p += __shfl_xor(p, off, 16);
        const float sc = (m[r] == -__builtin_inff()) ? 0.0f : __expf(m[r] - newm);
        l[r] = l[r] * sc + p;
        m[r] = newm;
      }
    }
  }

  if (ln == 0) {
#pragma unroll
    for (int r = 0; r < 8; ++r) {
      const int row = qrow + r + 8 * half;
      rowm[h * kS + row] = m[r];
      rowl[h * kS + row] = l[r];
    }
  }
}

// ---------------- pass 2: O = softmax_masked(S) @ V ------------------------
// grid (S/64, H), 128 threads; p relaid C->A frag via LDS (b128 reads), then
// WMMA with V B-frags. Output written pre-swizzled (A-frag) for the Wo GEMM.

__global__ __launch_bounds__(128) void attn_acc(
    const __bf16* __restrict__ Q, const __bf16* __restrict__ K,
    const __bf16* __restrict__ V, const float* __restrict__ thrv,
    const float* __restrict__ rowm, const float* __restrict__ rowl,
    __bf16* __restrict__ attn_swz) {
  __shared__ __align__(16) __bf16 ps[4 * 512];  // per-wave p tile, A-frag order
  const int h   = blockIdx.y;
  const int q0  = blockIdx.x * 64;
  const int wid = threadIdx.x >> 5;
  const int lane = threadIdx.x & 31;
  const int half = lane >> 4, ln = lane & 15;
  const float thr = thrv[h];
  const int qrow  = q0 + wid * 16;
  const int qf    = qrow >> 4;

  const v16bf a0 = load_frag(Q + (((size_t)(h * 2 + 0) * 128 + qf) << 9));
  const v16bf a1 = load_frag(Q + (((size_t)(h * 2 + 1) * 128 + qf) << 9));

  float m[8], linv[8];
#pragma unroll
  for (int r = 0; r < 8; ++r) {
    const int row = qrow + r + 8 * half;
    m[r] = rowm[h * kS + row];
    const float l = rowl[h * kS + row];
    linv[r] = (l > 0.0f) ? (1.0f / l) : 0.0f;
  }

  v8f o[4];
  v8f zero = {};
#pragma unroll
  for (int j = 0; j < 4; ++j) o[j] = zero;

  for (int kb = 0; kb < kS; kb += 32) {
#pragma unroll
    for (int t = 0; t < 2; ++t) {
      const int kf = (kb >> 4) + t;
      const v16bf b0 = load_frag(K + (((size_t)(h * 2 + 0) * 128 + kf) << 9));
      const v16bf b1 = load_frag(K + (((size_t)(h * 2 + 1) * 128 + kf) << 9));
      v8f s = {};
      s = wmma_bf16(a0, b0, s);
      s = wmma_bf16(a1, b1, s);
#pragma unroll
      for (int r = 0; r < 8; ++r) {
        const float v = s[r] * kScale;
        const float p = (v >= thr) ? __expf(v - m[r]) : 0.0f;
        ps[wid * 512 + a_swz(r + 8 * half, t * 16 + ln)] = (__bf16)p;
      }
    }
    __syncthreads();
    const v16bf pa = load_frag(&ps[wid * 512]);
#pragma unroll
    for (int j = 0; j < 4; ++j) {
      const v16bf bv =
          load_frag(V + (((size_t)(h * 64 + (kb >> 5)) * 4 + j) << 9));
      o[j] = wmma_bf16(pa, bv, o[j]);
    }
    __syncthreads();
  }

  // epilogue: divide by l, write in A-frag order for the output projection
#pragma unroll
  for (int j = 0; j < 4; ++j) {
#pragma unroll
    for (int r = 0; r < 8; ++r) {
      const int row = qrow + r + 8 * half;
      const int col = h * kHD + j * 16 + ln;
      const float v = o[j][r] * linv[r];
      attn_swz[(((size_t)(col >> 5) * 128 + (row >> 4)) << 9) +
               a_swz(row & 15, col & 31)] = (__bf16)v;
    }
  }
}

// ---------------- host-side launch -----------------------------------------

extern "C" void kernel_launch(void* const* d_in, const int* in_sizes, int n_in,
                              void* d_out, int out_size, void* d_ws,
                              size_t ws_size, hipStream_t stream) {
  const float* x  = (const float*)d_in[0];
  const float* Wq = (const float*)d_in[1];
  const float* bq = (const float*)d_in[2];
  const float* Wk = (const float*)d_in[3];
  const float* bk = (const float*)d_in[4];
  const float* Wv = (const float*)d_in[5];
  const float* bv = (const float*)d_in[6];
  const float* Wo = (const float*)d_in[7];
  const float* bo = (const float*)d_in[8];

  char* ws = (char*)d_ws;
  size_t off = 0;
  auto take = [&](size_t bytes) { char* p = ws + off; off += bytes; return p; };
  __bf16* xb   = (__bf16*)take((size_t)kS * kD * 2);        // A-frag order
  __bf16* wqb  = (__bf16*)take((size_t)kD * kD * 2);        // B-frag order
  __bf16* wkb  = (__bf16*)take((size_t)kD * kD * 2);
  __bf16* wvb  = (__bf16*)take((size_t)kD * kD * 2);
  __bf16* wob  = (__bf16*)take((size_t)kD * kD * 2);
  __bf16* Qb   = (__bf16*)take((size_t)kH * kS * kHD * 2);  // A-frag order
  __bf16* Kb   = (__bf16*)take((size_t)kH * kS * kHD * 2);  // B-frag (K^T)
  __bf16* Vb   = (__bf16*)take((size_t)kH * kS * kHD * 2);  // B-frag order
  __bf16* attn = (__bf16*)take((size_t)kS * kD * 2);        // A-frag order
  unsigned int* hist = (unsigned int*)take((size_t)kH * NBINS * 4);
  float* thr  = (float*)take(256);
  float* rowm = (float*)take((size_t)kH * kS * 4);
  float* rowl = (float*)take((size_t)kH * kS * 4);

  const int nx = kS * kD, nw = kD * kD;
  cvt_x_swzA<<<(nx + 255) / 256, 256, 0, stream>>>(x, xb);
  cvt_w_swzB<<<(nw + 255) / 256, 256, 0, stream>>>(Wq, wqb);
  cvt_w_swzB<<<(nw + 255) / 256, 256, 0, stream>>>(Wk, wkb);
  cvt_w_swzB<<<(nw + 255) / 256, 256, 0, stream>>>(Wv, wvb);
  cvt_w_swzB<<<(nw + 255) / 256, 256, 0, stream>>>(Wo, wob);

  dim3 gg(kS / 128, kD / 128);
  gemm_swz<<<gg, 256, 0, stream>>>(xb, wqb, bq, Qb, nullptr, 0);
  gemm_swz<<<gg, 256, 0, stream>>>(xb, wkb, bk, Kb, nullptr, 1);
  gemm_swz<<<gg, 256, 0, stream>>>(xb, wvb, bv, Vb, nullptr, 2);

  hipMemsetAsync(hist, 0, (size_t)kH * NBINS * 4, stream);
  hist_scores<<<dim3(kS / 64, kS / 64, kH), 256, 0, stream>>>(Qb, Kb, hist);
  find_thr<<<kH, 32, 0, stream>>>(hist, thr);

  row_stats<<<dim3(kS / 64, kH), 128, 0, stream>>>(Qb, Kb, thr, rowm, rowl);
  attn_acc<<<dim3(kS / 64, kH), 128, 0, stream>>>(Qb, Kb, Vb, thr, rowm, rowl,
                                                  attn);

  gemm_swz<<<gg, 256, 0, stream>>>(attn, wob, bo, nullptr, (float*)d_out, 3);
}
